// SelfAttentionSplitted_34497177321766
// MI455X (gfx1250) — matrix operand order repaired and checked by reading, not compiled, not run on hardware
//
#include <hip/hip_runtime.h>

// ---------------------------------------------------------------------------
// MI455X (gfx1250) self-attention:  bf16 WMMA GEMMs + flash attention,
// with CDNA5 async global->LDS tile staging (ASYNCcnt) and double buffering.
// Shapes: B=4, T=2048, K=1024, H=16, HD=64.
// ---------------------------------------------------------------------------

typedef __bf16 bf16;
typedef __attribute__((ext_vector_type(8)))  bf16  bf16x8;
typedef __attribute__((ext_vector_type(16))) bf16  bf16x16;
typedef __attribute__((ext_vector_type(8)))  float f32x8;

#define BATCH 4
#define SEQ   2048
#define KDIM  1024
#define HEADS 16
#define HD    64

// Build a 16-element bf16 A/B fragment from two 16-byte chunks.
static __device__ inline bf16x16 frag2(const bf16* lo, const bf16* hi) {
  bf16x8 l = *(const bf16x8*)lo;
  bf16x8 h = *(const bf16x8*)hi;
  bf16x16 r;
#pragma unroll
  for (int i = 0; i < 8; ++i) { r[i] = l[i]; r[i + 8] = h[i]; }
  return r;
}

// Low 32 bits of a generic pointer into __shared__ == LDS byte offset
// (flat LDS aperture keeps the LDS address in addr[31:0] per CDNA5 ISA).
static __device__ inline unsigned lds_off(const void* p) {
  return (unsigned)(unsigned long long)p;
}

// CDNA5 async copy: 16B per lane, global -> LDS, tracked by ASYNCcnt.
static __device__ inline void async_ld_b128(unsigned ldsoff, const void* g) {
  asm volatile("global_load_async_to_lds_b128 %0, %1, off"
               :: "v"(ldsoff), "v"((unsigned long long)g)
               : "memory");
}

static __device__ inline void wait_async0() {
  asm volatile("s_wait_asynccnt 0" ::: "memory");
}

// ---------------------------------------------------------------------------
// fp32 -> bf16 conversion
// ---------------------------------------------------------------------------
__global__ void cvt_f32_bf16(const float* __restrict__ in,
                             bf16* __restrict__ out, long n) {
  long i = (long)blockIdx.x * blockDim.x + threadIdx.x;
  long stride = (long)gridDim.x * blockDim.x;
  for (; i < n; i += stride) out[i] = (bf16)in[i];
}

// ---------------------------------------------------------------------------
// Tiled WMMA GEMM:  C[m,n] = scale * sum_k A[m,k] * W[n,k]  (+ bias[n])
// A: [M,Kd] bf16 row-major,  W: [N,Kd] bf16 row-major (x @ W^T: W rows are
// directly the column-major B operand).
// Block: 256 threads = 8 waves, 128(M) x 64(N) tile, K-step 32,
// double-buffered async LDS staging overlapped with WMMA compute.
// ---------------------------------------------------------------------------
#define GBM  128
#define GBN  64
#define GBK  32
#define GAST 48   // padded LDS k-stride (96B, 16B-aligned chunks)
#define GWST 48

template <bool OUT_F32>
__global__ __launch_bounds__(256)
void gemm_bf16_wmma(const bf16* __restrict__ A, const bf16* __restrict__ W,
                    void* __restrict__ Cout, const float* __restrict__ bias,
                    int M, int N, int Kd, float scale) {
  __shared__ bf16 As[2 * GBM * GAST];   // 24 KB (double buffered)
  __shared__ bf16 Ws[2 * GBN * GWST];   // 12 KB

  const int tid  = threadIdx.x;
  const int lane = tid & 31;
  const int wave = tid >> 5;
  const int l15  = lane & 15;
  const int lh   = lane >> 4;       // 0: lanes 0-15, 1: lanes 16-31

  const int m0 = blockIdx.y * GBM;
  const int n0 = blockIdx.x * GBN;
  const int wm = (wave & 3) * 32;   // 4 waves along M
  const int wn = (wave >> 2) * 32;  // 2 waves along N

  // Issue one tile's async copies (A: 2 chunks/thread, W: 1 chunk/thread).
  auto stage = [&](int k0, int buf) {
    bf16* Asb = &As[buf * GBM * GAST];
    bf16* Wsb = &Ws[buf * GBN * GWST];
#pragma unroll
    for (int j = 0; j < 2; ++j) {
      int c   = tid + j * 256;
      int row = c >> 2;
      int col = (c & 3) * 8;
      async_ld_b128(lds_off(&Asb[row * GAST + col]),
                    &A[(size_t)(m0 + row) * Kd + k0 + col]);
    }
    {
      int row = tid >> 2;
      int col = (tid & 3) * 8;
      async_ld_b128(lds_off(&Wsb[row * GWST + col]),
                    &W[(size_t)(n0 + row) * Kd + k0 + col]);
    }
  };

  f32x8 acc[2][2] = {};
  const int nk = Kd / GBK;

  stage(0, 0);
  for (int it = 0; it < nk; ++it) {
    const int cur = it & 1;
    wait_async0();        // this wave's share of tile `it` is in LDS
    __syncthreads();      // every wave's share is in LDS; prev compute done
    if (it + 1 < nk) stage((it + 1) * GBK, cur ^ 1);  // overlap with WMMA
    if (it + 2 < nk)      // light lookahead prefetch of the A panel
      __builtin_prefetch(&A[(size_t)(m0 + (tid >> 1)) * Kd +
                            (it + 2) * GBK + (tid & 1) * 16], 0, 0);

    const bf16* Asb = &As[cur * GBM * GAST];
    const bf16* Wsb = &Ws[cur * GBN * GWST];

    // A operand: lane<16 -> K {0..7,16..23}, lane>=16 -> K {8..15,24..31}
    bf16x16 af[2];
#pragma unroll
    for (int ms = 0; ms < 2; ++ms) {
      const bf16* p = &Asb[(wm + ms * 16 + l15) * GAST + lh * 8];
      af[ms] = frag2(p, p + 16);
    }
    // B operand: lane<16 -> K {0..7,16..23}, lane>=16 -> K {8..15,24..31}
    bf16x16 bfr[2];
#pragma unroll
    for (int ns = 0; ns < 2; ++ns) {
      const bf16* p = &Wsb[(wn + ns * 16 + l15) * GWST + lh * 8];
      bfr[ns] = frag2(p, p + 16);
    }
#pragma unroll
    for (int ms = 0; ms < 2; ++ms)
#pragma unroll
      for (int ns = 0; ns < 2; ++ns)
        acc[ms][ns] = __builtin_amdgcn_wmma_f32_16x16x32_bf16(
            false, af[ms], false, bfr[ns], (short)0, acc[ms][ns], false, false);
  }

  // Epilogue. C/D layout: lane = column, VGPR i = row (i + 8*lh within 16).
#pragma unroll
  for (int ms = 0; ms < 2; ++ms) {
#pragma unroll
    for (int ns = 0; ns < 2; ++ns) {
      int colg = n0 + wn + ns * 16 + l15;
      int rowg = m0 + wm + ms * 16 + lh * 8;
      if (OUT_F32) {
        float bb = bias ? bias[colg] : 0.f;
        float* o = (float*)Cout;
#pragma unroll
        for (int i = 0; i < 8; ++i)
          o[(size_t)(rowg + i) * N + colg] = acc[ms][ns][i] * scale + bb;
      } else {
        bf16* o = (bf16*)Cout;
#pragma unroll
        for (int i = 0; i < 8; ++i)
          o[(size_t)(rowg + i) * N + colg] = (bf16)(acc[ms][ns][i] * scale);
      }
    }
  }
}

// ---------------------------------------------------------------------------
// Flash attention, one (b, h) per blockIdx.{z,y}; 8 waves x 32 q-rows each.
// K tile async-copied to LDS (double buffered); V loaded through VGPRs and
// transposed into LDS. Online softmax with additive -10000 padding penalty.
// ---------------------------------------------------------------------------
#define KST 80   // padded LDS stride (160B, multiple of 16B)

__global__ __launch_bounds__(256)
void attn_flash_wmma(const bf16* __restrict__ Q, const bf16* __restrict__ Kc,
                     const bf16* __restrict__ V, bf16* __restrict__ O,
                     const long long* __restrict__ lengths) {
  __shared__ bf16 Kl[2 * 64 * KST];    // [buf][key][hd]     20 KB
  __shared__ bf16 Vt[2 * 64 * KST];    // [buf][hd][key]     20 KB
  __shared__ bf16 Pl[8 * 32 * KST];    // per-wave P tile    40 KB

  const int tid  = threadIdx.x;
  const int lane = tid & 31;
  const int wave = tid >> 5;
  const int l15  = lane & 15;
  const int lh   = lane >> 4;

  const int b   = blockIdx.z;
  const int h   = blockIdx.y;
  const int q0  = blockIdx.x * 256 + wave * 32;
  const int len = (int)lengths[b];

  // Stage K (async -> LDS) and V (VGPR transpose -> LDS) for one 64-key tile.
  auto stage_kv = [&](int tile, int buf) {
    bf16* Kb_ = &Kl[buf * 64 * KST];
    bf16* Vb_ = &Vt[buf * 64 * KST];
    const int kb = tile * 64;
#pragma unroll
    for (int j = 0; j < 2; ++j) {
      int c   = tid + j * 256;
      int row = c >> 3;          // key within tile
      int col = (c & 7) * 8;     // hd offset
      size_t g = (size_t)(b * SEQ + kb + row) * KDIM + h * HD + col;
      async_ld_b128(lds_off(&Kb_[row * KST + col]), &Kc[g]);
      bf16x8 v = *(const bf16x8*)&V[g];
#pragma unroll
      for (int jj = 0; jj < 8; ++jj) Vb_[(col + jj) * KST + row] = v[jj];
    }
  };

  // Preload Q fragments for this wave's 32 rows (2 m-frags x 2 k-steps).
  bf16x16 aq[2][2];
#pragma unroll
  for (int ms = 0; ms < 2; ++ms)
#pragma unroll
    for (int ks = 0; ks < 2; ++ks) {
      const bf16* p = &Q[(size_t)(b * SEQ + q0 + ms * 16 + l15) * KDIM +
                         h * HD + ks * 32 + lh * 8];
      aq[ms][ks] = frag2(p, p + 16);
    }

  f32x8 o[2][4] = {};
  float mrow[2][8], lrow[2][8];
#pragma unroll
  for (int ms = 0; ms < 2; ++ms)
#pragma unroll
    for (int i = 0; i < 8; ++i) { mrow[ms][i] = -3.0e38f; lrow[ms][i] = 0.f; }

  bf16* Pw = &Pl[wave * 32 * KST];
  const int nt = SEQ / 64;

  stage_kv(0, 0);
  for (int it = 0; it < nt; ++it) {
    const int cur = it & 1;
    const int kb  = it * 64;
    wait_async0();
    __syncthreads();
    if (it + 1 < nt) stage_kv(it + 1, cur ^ 1);   // overlap with compute

    const bf16* Kb_ = &Kl[cur * 64 * KST];
    const bf16* Vb_ = &Vt[cur * 64 * KST];

    // S = Q * K^T : 2 m-frags x 4 key-subtiles, hd split into 2 k-steps.
    f32x8 s[2][4] = {};
#pragma unroll
    for (int ks = 0; ks < 2; ++ks) {
      bf16x16 bk[4];
#pragma unroll
      for (int ns = 0; ns < 4; ++ns) {
        const bf16* p = &Kb_[(ns * 16 + l15) * KST + ks * 32 + lh * 8];
        bk[ns] = frag2(p, p + 16);
      }
#pragma unroll
      for (int ms = 0; ms < 2; ++ms)
#pragma unroll
        for (int ns = 0; ns < 4; ++ns)
          s[ms][ns] = __builtin_amdgcn_wmma_f32_16x16x32_bf16(
              false, aq[ms][ks], false, bk[ns], (short)0, s[ms][ns],
              false, false);
    }

    // Additive padding mask + running row max (reduce across 16-lane half).
    float mnew[2][8];
#pragma unroll
    for (int ms = 0; ms < 2; ++ms)
#pragma unroll
      for (int i = 0; i < 8; ++i) mnew[ms][i] = mrow[ms][i];
#pragma unroll
    for (int ms = 0; ms < 2; ++ms)
#pragma unroll
      for (int ns = 0; ns < 4; ++ns) {
        int   col = kb + ns * 16 + l15;
        float pen = (col < len) ? 0.f : -10000.f;
#pragma unroll
        for (int i = 0; i < 8; ++i) {
          float vv = s[ms][ns][i] + pen;
          s[ms][ns][i] = vv;
          mnew[ms][i] = fmaxf(mnew[ms][i], vv);
        }
      }
#pragma unroll
    for (int ms = 0; ms < 2; ++ms)
#pragma unroll
      for (int i = 0; i < 8; ++i)
#pragma unroll
        for (int off = 1; off < 16; off <<= 1)
          mnew[ms][i] = fmaxf(mnew[ms][i], __shfl_xor(mnew[ms][i], off, 32));

    float resc[2][8], rsum[2][8];
#pragma unroll
    for (int ms = 0; ms < 2; ++ms)
#pragma unroll
      for (int i = 0; i < 8; ++i) {
        resc[ms][i] = __expf(mrow[ms][i] - mnew[ms][i]);
        mrow[ms][i] = mnew[ms][i];
        rsum[ms][i] = 0.f;
      }
    // P = exp(S - m), spill to per-wave LDS (accumulator->A-operand transpose)
#pragma unroll
    for (int ms = 0; ms < 2; ++ms)
#pragma unroll
      for (int ns = 0; ns < 4; ++ns)
#pragma unroll
        for (int i = 0; i < 8; ++i) {
          float p = __expf(s[ms][ns][i] - mnew[ms][i]);
          rsum[ms][i] += p;
          Pw[(ms * 16 + i + lh * 8) * KST + ns * 16 + l15] = (bf16)p;
        }
#pragma unroll
    for (int ms = 0; ms < 2; ++ms)
#pragma unroll
      for (int i = 0; i < 8; ++i) {
#pragma unroll
        for (int off = 1; off < 16; off <<= 1)
          rsum[ms][i] += __shfl_xor(rsum[ms][i], off, 32);
        lrow[ms][i] = lrow[ms][i] * resc[ms][i] + rsum[ms][i];
      }
#pragma unroll
    for (int ms = 0; ms < 2; ++ms)
#pragma unroll
      for (int ns = 0; ns < 4; ++ns)
#pragma unroll
        for (int i = 0; i < 8; ++i) o[ms][ns][i] *= resc[ms][i];

    // Make this wave's LDS P stores visible to its own ds loads.
    asm volatile("s_wait_dscnt 0" ::: "memory");

    // O += P * V  (keys are the K dimension: two 32-deep steps)
#pragma unroll
    for (int ks = 0; ks < 2; ++ks) {
      bf16x16 ap[2], bv[4];
#pragma unroll
      for (int ms = 0; ms < 2; ++ms) {
        const bf16* p = &Pw[(ms * 16 + l15) * KST + ks * 32 + lh * 8];
        ap[ms] = frag2(p, p + 16);
      }
#pragma unroll
      for (int ns = 0; ns < 4; ++ns) {
        const bf16* p = &Vb_[(ns * 16 + l15) * KST + ks * 32 + lh * 8];
        bv[ns] = frag2(p, p + 16);
      }
#pragma unroll
      for (int ms = 0; ms < 2; ++ms)
#pragma unroll
        for (int ns = 0; ns < 4; ++ns)
          o[ms][ns] = __builtin_amdgcn_wmma_f32_16x16x32_bf16(
              false, ap[ms], false, bv[ns], (short)0, o[ms][ns], false, false);
    }
  }

  // Normalize by row sums and store bf16 attention output.
#pragma unroll
  for (int ms = 0; ms < 2; ++ms)
#pragma unroll
    for (int ns = 0; ns < 4; ++ns)
#pragma unroll
      for (int i = 0; i < 8; ++i) {
        int t = q0 + ms * 16 + i + lh * 8;
        int d = h * HD + ns * 16 + l15;
        O[(size_t)(b * SEQ + t) * KDIM + d] =
            (bf16)(o[ms][ns][i] / lrow[ms][i]);
      }
}

// ---------------------------------------------------------------------------
// Host-side launch
// ---------------------------------------------------------------------------
extern "C" void kernel_launch(void* const* d_in, const int* in_sizes, int n_in,
                              void* d_out, int out_size, void* d_ws,
                              size_t ws_size, hipStream_t stream) {
  (void)in_sizes; (void)n_in; (void)out_size; (void)ws_size;

  const float*     x0      = (const float*)d_in[0];
  const long long* lengths = (const long long*)d_in[1];
  const float*     Wq      = (const float*)d_in[2];
  const float*     Wk      = (const float*)d_in[3];
  const float*     Wv      = (const float*)d_in[4];
  const float*     Wu      = (const float*)d_in[5];
  const float*     bu      = (const float*)d_in[6];
  float*           out     = (float*)d_out;

  const size_t NX = (size_t)BATCH * SEQ * KDIM;  // 8 Mi elements
  const size_t NW = (size_t)KDIM * KDIM;         // 1 Mi elements

  // Workspace layout (bf16): x | Wq Wk Wv Wu | Q | K | V | O  (~88 MB)
  bf16* xb  = (bf16*)d_ws;
  bf16* wqb = xb  + NX;
  bf16* wkb = wqb + NW;
  bf16* wvb = wkb + NW;
  bf16* wub = wvb + NW;
  bf16* Qb  = wub + NW;
  bf16* Kb  = Qb  + NX;
  bf16* Vb  = Kb  + NX;
  bf16* Ob  = Vb  + NX;

  cvt_f32_bf16<<<4096, 256, 0, stream>>>(x0, xb, (long)NX);
  cvt_f32_bf16<<<1024, 256, 0, stream>>>(Wq, wqb, (long)NW);
  cvt_f32_bf16<<<1024, 256, 0, stream>>>(Wk, wkb, (long)NW);
  cvt_f32_bf16<<<1024, 256, 0, stream>>>(Wv, wvb, (long)NW);
  cvt_f32_bf16<<<1024, 256, 0, stream>>>(Wu, wub, (long)NW);

  dim3 gg(KDIM / GBN, (BATCH * SEQ) / GBM);   // 16 x 64 blocks
  const float qs = 0.35355339059327373f;      // HD^(-1/4), applied to Q and K

  gemm_bf16_wmma<false><<<gg, 256, 0, stream>>>(xb, wqb, Qb, nullptr,
                                                BATCH * SEQ, KDIM, KDIM, qs);
  gemm_bf16_wmma<false><<<gg, 256, 0, stream>>>(xb, wkb, Kb, nullptr,
                                                BATCH * SEQ, KDIM, KDIM, qs);
  gemm_bf16_wmma<false><<<gg, 256, 0, stream>>>(xb, wvb, Vb, nullptr,
                                                BATCH * SEQ, KDIM, KDIM, 1.0f);

  attn_flash_wmma<<<dim3(SEQ / 256, HEADS, BATCH), 256, 0, stream>>>(
      Qb, Kb, Vb, Ob, lengths);

  gemm_bf16_wmma<true><<<gg, 256, 0, stream>>>(Ob, wub, out, bu,
                                               BATCH * SEQ, KDIM, KDIM, 1.0f);
}
